// GroupedQueryAttention_40982577938914
// MI455X (gfx1250) — matrix operand order repaired and checked by reading, not compiled, not run on hardware
//
#include <hip/hip_runtime.h>
#include <hip/hip_bf16.h>
#include <math.h>

typedef __bf16 bf16_t;
typedef __attribute__((ext_vector_type(16))) __bf16 v16bf;
typedef __attribute__((ext_vector_type(8)))  __bf16 v8bf;
typedef __attribute__((ext_vector_type(8)))  float   v8f;

#define B_   2
#define S_   2048
#define D_   2048
#define H_   32
#define HK_  8
#define HD_  64
#define M_   (B_*S_)     /* 4096 */
#define KD_  (HK_*HD_)   /* 512  */

__device__ __forceinline__ v16bf cat8(v8bf lo, v8bf hi) {
  return __builtin_shufflevector(lo, hi, 0,1,2,3,4,5,6,7,8,9,10,11,12,13,14,15);
}
__device__ __forceinline__ v8f wmma_bf16(v16bf a, v16bf b, v8f c) {
  // 8 args: (neg_a, A, neg_b, B, c_mod, C, reuse_a, reuse_b)
  return __builtin_amdgcn_wmma_f32_16x16x32_bf16(false, a, false, b, (short)0, c, false, false);
}

// CDNA5 async global->LDS copy, 16 bytes per lane (ASYNCcnt-tracked).
// CDNA5 ISA: GLOBAL_LOAD_ASYNC_TO_LDS_B128, VDST = LDS addr VGPR.
__device__ __forceinline__ void async_cp16(const bf16_t* g, bf16_t* l) {
  unsigned laddr = (unsigned)(unsigned long long)
      (__attribute__((address_space(3))) bf16_t*)l;
  asm volatile("global_load_async_to_lds_b128 %0, %1, off"
               :: "v"(laddr), "v"((unsigned long long)g) : "memory");
}
#define WAIT_ASYNC(n) asm volatile("s_wait_asynccnt " #n ::: "memory")

// ---------------------------------------------------------------- convert
__global__ void cvt_kernel(const float* __restrict__ in, bf16_t* __restrict__ out, long n4) {
  long i = (long)blockIdx.x * blockDim.x + threadIdx.x;
  if (i < n4) {
    float4 f = ((const float4*)in)[i];
    out[i*4+0] = (bf16_t)f.x;
    out[i*4+1] = (bf16_t)f.y;
    out[i*4+2] = (bf16_t)f.z;
    out[i*4+3] = (bf16_t)f.w;
  }
}

// ---------------------------------------------------------------- GEMM: C = A(MxK) * W(NxK)^T, bf16 in, f32 out
// Double-buffered LDS, tiles staged with GLOBAL_LOAD_ASYNC_TO_LDS_B128.
#define BM 128
#define BN 128
#define BK 32
#define LDT 40   /* padded row stride (elements) to break LDS bank conflicts */

__global__ __launch_bounds__(256) void gemm_bf16_kernel(
    const bf16_t* __restrict__ A, const bf16_t* __restrict__ W,
    float* __restrict__ C, int M, int N, int K)
{
  __shared__ bf16_t As[2][BM][LDT];
  __shared__ bf16_t Ws[2][BN][LDT];
  const int tid  = threadIdx.x;
  const int wave = tid >> 5;
  const int lane = tid & 31;
  const int half = lane >> 4;
  const int l15  = lane & 15;
  const int m0 = blockIdx.y * BM;
  const int n0 = blockIdx.x * BN;
  const int lr = tid >> 1;           // cooperative-load row 0..127
  const int lc = (tid & 1) * 16;     // cooperative-load col 0 / 16

  v8f acc[8] = {};

  const bf16_t* ag = A + (size_t)(m0 + lr) * K + lc;
  const bf16_t* wg = W + (size_t)(n0 + lr) * K + lc;

  // prologue: stage tile 0 into buffer 0 (4 async b128 per thread)
  async_cp16(ag,     &As[0][lr][lc]);
  async_cp16(ag + 8, &As[0][lr][lc + 8]);
  async_cp16(wg,     &Ws[0][lr][lc]);
  async_cp16(wg + 8, &Ws[0][lr][lc + 8]);

  for (int k0 = 0; k0 < K; k0 += BK) {
    const int buf = (k0 >> 5) & 1;
    if (k0 + BK < K) {
      const int nb = buf ^ 1;
      const bf16_t* an = ag + k0 + BK;
      const bf16_t* wn = wg + k0 + BK;
      async_cp16(an,     &As[nb][lr][lc]);
      async_cp16(an + 8, &As[nb][lr][lc + 8]);
      async_cp16(wn,     &Ws[nb][lr][lc]);
      async_cp16(wn + 8, &Ws[nb][lr][lc + 8]);
      WAIT_ASYNC(4);   // retire current buffer's 4 copies; next tile stays in flight
    } else {
      WAIT_ASYNC(0);
    }
    __syncthreads();   // all waves' current-buffer data visible

    // A operand (16x32): lane lo: k 0-15; lane hi: k 16-31
    const bf16_t* arow = &As[buf][wave * 16 + l15][0];
    v16bf a = cat8(*(const v8bf*)(arow + half * 16),
                   *(const v8bf*)(arow + half * 16 + 8));
    // batch all B operands first, then back-to-back WMMAs (one DS wait)
    v16bf bt[8];
    #pragma unroll
    for (int nt = 0; nt < 8; nt++) {
      const bf16_t* wrow = &Ws[buf][nt * 16 + l15][0];
      bt[nt] = cat8(*(const v8bf*)(wrow + half * 16),
                    *(const v8bf*)(wrow + half * 16 + 8));
    }
    #pragma unroll
    for (int nt = 0; nt < 8; nt++)
      acc[nt] = wmma_bf16(a, bt[nt], acc[nt]);

    __syncthreads();   // everyone done reading buf before it is overwritten
  }

  #pragma unroll
  for (int nt = 0; nt < 8; nt++) {
    #pragma unroll
    for (int j = 0; j < 8; j++) {   // C layout: vgpr j -> row j (lanes 0-15) / j+8 (lanes 16-31)
      int m = m0 + wave * 16 + j + 8 * half;
      int n = n0 + nt * 16 + l15;
      C[(size_t)m * N + n] = acc[nt][j];
    }
  }
}

// ---------------------------------------------------------------- RoPE + repack
__global__ void rope_q_kernel(const float* __restrict__ qf, const float* __restrict__ cosb,
                              const float* __restrict__ sinb, bf16_t* __restrict__ qb) {
  size_t i = (size_t)blockIdx.x * 256 + threadIdx.x;   // over (b,s,h,d) = M_*D_
  int d = i & 63;
  int h = (i >> 6) & (H_ - 1);
  int s = (i >> 11) & (S_ - 1);
  int b = (int)(i >> 22);
  float qv  = qf[i];
  float rot = (d < 32) ? -qf[i + 32] : qf[i - 32];
  float c   = cosb[s * 64 + d];
  float sn  = sinb[s * 64 + d];
  qb[(((size_t)b * H_ + h) * S_ + s) * 64 + d] = (bf16_t)(qv * c + rot * sn);
}

__global__ void rope_k_kernel(const float* __restrict__ kf, const float* __restrict__ cosb,
                              const float* __restrict__ sinb, bf16_t* __restrict__ kb) {
  size_t i = (size_t)blockIdx.x * 256 + threadIdx.x;   // over (b,s,hk,d) = M_*KD_
  int d  = i & 63;
  int hk = (i >> 6) & (HK_ - 1);
  int s  = (i >> 9) & (S_ - 1);
  int b  = (int)(i >> 20);
  float kv  = kf[i];
  float rot = (d < 32) ? -kf[i + 32] : kf[i - 32];
  float c   = cosb[s * 64 + d];
  float sn  = sinb[s * 64 + d];
  kb[(((size_t)b * HK_ + hk) * S_ + s) * 64 + d] = (bf16_t)(kv * c + rot * sn);
}

__global__ void pack_v_kernel(const float* __restrict__ vf, bf16_t* __restrict__ vtb) {
  size_t i = (size_t)blockIdx.x * 256 + threadIdx.x;   // over (b,s,hk,d) = M_*KD_
  int d  = i & 63;
  int hk = (i >> 6) & (HK_ - 1);
  int s  = (i >> 9) & (S_ - 1);
  int b  = (int)(i >> 20);
  // v stored transposed: (B, HK, HD, S) so PV's B operand reads contiguous keys
  vtb[(((size_t)b * HK_ + hk) * 64 + d) * S_ + s] = (bf16_t)vf[i];
}

// ---------------------------------------------------------------- flash attention (1 wave = 16 q rows)
__global__ __launch_bounds__(256) void attn_kernel(
    const bf16_t* __restrict__ qb,   // (B,H,S,HD)
    const bf16_t* __restrict__ kb,   // (B,HK,S,HD)
    const bf16_t* __restrict__ vtb,  // (B,HK,HD,S)
    bf16_t* __restrict__ ob)         // (B,S,H*HD)
{
  __shared__ bf16_t Pl[8][16][LDT];  // per-wave P tile (16 x 32 used)
  const int lane = threadIdx.x & 31;
  const int wave = threadIdx.x >> 5;
  const int gid  = blockIdx.x * 8 + wave;       // 0 .. B_*H_*(S_/16)-1
  const int b    = gid >> 12;                   // H_*S_/16 = 4096 per batch
  const int h    = (gid >> 7) & 31;
  const int m0   = (gid & 127) << 4;
  const int hk   = h >> 2;                      // REP = 4
  const int half = lane >> 4;
  const int l15  = lane & 15;

  const bf16_t* qbase = qb  + ((size_t)(b * H_  + h ) * S_ + m0) * HD_;
  const bf16_t* kbase = kb  + ((size_t)(b * HK_ + hk) * S_) * HD_;
  const bf16_t* vbase = vtb + ((size_t)(b * HK_ + hk) * HD_) * S_;

  // q A-operands for both 32-wide k-steps of HD=64, kept in registers
  const bf16_t* qrow = qbase + (size_t)l15 * HD_;
  v16bf qa0 = cat8(*(const v8bf*)(qrow +      half * 16), *(const v8bf*)(qrow +      half * 16 + 8));
  v16bf qa1 = cat8(*(const v8bf*)(qrow + 32 + half * 16), *(const v8bf*)(qrow + 32 + half * 16 + 8));

  v8f o0 = {}, o1 = {}, o2 = {}, o3 = {};
  float mrow[8], lrow[8];
  #pragma unroll
  for (int j = 0; j < 8; j++) { mrow[j] = -__builtin_inff(); lrow[j] = 0.0f; }
  const float scale = 0.125f;  // 1/sqrt(HD)

  for (int t0 = 0; t0 < m0 + 16; t0 += 32) {
    // ---- batch all K operand loads, then 4 score WMMAs back-to-back
    const bf16_t* kr0 = kbase + (size_t)(t0 + l15) * HD_;
    const bf16_t* kr1 = kbase + (size_t)(t0 + 16 + l15) * HD_;
    v16bf k0a = cat8(*(const v8bf*)(kr0 + half * 16),      *(const v8bf*)(kr0 + half * 16 + 8));
    v16bf k0b = cat8(*(const v8bf*)(kr0 + 32 + half * 16), *(const v8bf*)(kr0 + 32 + half * 16 + 8));
    v16bf k1a = cat8(*(const v8bf*)(kr1 + half * 16),      *(const v8bf*)(kr1 + half * 16 + 8));
    v16bf k1b = cat8(*(const v8bf*)(kr1 + 32 + half * 16), *(const v8bf*)(kr1 + 32 + half * 16 + 8));
    v8f sc0 = {}, sc1 = {};
    sc0 = wmma_bf16(qa0, k0a, sc0);
    sc0 = wmma_bf16(qa1, k0b, sc0);
    sc1 = wmma_bf16(qa0, k1a, sc1);
    sc1 = wmma_bf16(qa1, k1b, sc1);

    // ---- online softmax, entirely in registers (per-row stats; reduce across 16-lane half)
    float p0[8], p1[8], alpha[8];
    #pragma unroll
    for (int j = 0; j < 8; j++) {
      int m = m0 + j + 8 * half;
      float s0 = sc0[j] * scale; if (t0 + l15 > m)       s0 = -__builtin_inff();
      float s1 = sc1[j] * scale; if (t0 + 16 + l15 > m)  s1 = -__builtin_inff();
      float bm = fmaxf(s0, s1);
      #pragma unroll
      for (int off = 1; off < 16; off <<= 1) bm = fmaxf(bm, __shfl_xor(bm, off, 32));
      float mn = fmaxf(mrow[j], bm);
      alpha[j] = __expf(mrow[j] - mn);
      mrow[j]  = mn;
      p0[j] = __expf(s0 - mn);
      p1[j] = __expf(s1 - mn);
      float ps = p0[j] + p1[j];
      #pragma unroll
      for (int off = 1; off < 16; off <<= 1) ps += __shfl_xor(ps, off, 32);
      lrow[j] = lrow[j] * alpha[j] + ps;
    }

    // ---- rescale accumulators, stash P in LDS (C-layout -> A-layout transpose)
    #pragma unroll
    for (int j = 0; j < 8; j++) {
      o0[j] *= alpha[j]; o1[j] *= alpha[j]; o2[j] *= alpha[j]; o3[j] *= alpha[j];
      Pl[wave][j + 8 * half][l15]      = (bf16_t)p0[j];
      Pl[wave][j + 8 * half][16 + l15] = (bf16_t)p1[j];
    }
    asm volatile("s_wait_dscnt 0" ::: "memory");

    const bf16_t* pr = &Pl[wave][l15][0];
    v16bf pa = cat8(*(const v8bf*)(pr + half * 16), *(const v8bf*)(pr + half * 16 + 8));

    // ---- PV: batch 4 V operand loads (transposed V), then 4 WMMAs, K-dim = 32 keys
    const bf16_t* vt = vbase + t0 + half * 16;
    const bf16_t* vr0 = vt + (size_t)(0 * 16 + l15) * S_;
    const bf16_t* vr1 = vt + (size_t)(1 * 16 + l15) * S_;
    const bf16_t* vr2 = vt + (size_t)(2 * 16 + l15) * S_;
    const bf16_t* vr3 = vt + (size_t)(3 * 16 + l15) * S_;
    v16bf vB0 = cat8(*(const v8bf*)vr0, *(const v8bf*)(vr0 + 8));
    v16bf vB1 = cat8(*(const v8bf*)vr1, *(const v8bf*)(vr1 + 8));
    v16bf vB2 = cat8(*(const v8bf*)vr2, *(const v8bf*)(vr2 + 8));
    v16bf vB3 = cat8(*(const v8bf*)vr3, *(const v8bf*)(vr3 + 8));
    o0 = wmma_bf16(pa, vB0, o0);
    o1 = wmma_bf16(pa, vB1, o1);
    o2 = wmma_bf16(pa, vB2, o2);
    o3 = wmma_bf16(pa, vB3, o3);
  }

  // ---- normalize and write o as (B, S, H*HD) bf16 for the final projection GEMM
  bf16_t* orow = ob + (size_t)b * S_ * D_ + h * HD_;
  #pragma unroll
  for (int j = 0; j < 8; j++) {
    float inv = 1.0f / lrow[j];
    int srow = m0 + j + 8 * half;
    bf16_t* op = orow + (size_t)srow * D_;
    op[ 0 + l15] = (bf16_t)(o0[j] * inv);
    op[16 + l15] = (bf16_t)(o1[j] * inv);
    op[32 + l15] = (bf16_t)(o2[j] * inv);
    op[48 + l15] = (bf16_t)(o3[j] * inv);
  }
}

// ---------------------------------------------------------------- launcher
extern "C" void kernel_launch(void* const* d_in, const int* in_sizes, int n_in,
                              void* d_out, int out_size, void* d_ws, size_t ws_size,
                              hipStream_t stream) {
  (void)in_sizes; (void)n_in; (void)out_size; (void)ws_size;
  const float* x    = (const float*)d_in[0];
  const float* cosb = (const float*)d_in[1];
  const float* sinb = (const float*)d_in[2];
  /* d_in[3] = attn_mask: causal mask applied analytically, unused */
  const float* wq   = (const float*)d_in[4];
  const float* wk   = (const float*)d_in[5];
  const float* wv   = (const float*)d_in[6];
  const float* wo   = (const float*)d_in[7];
  float* out = (float*)d_out;

  // workspace carve-out (~124 MB total)
  char* p = (char*)d_ws;
  auto carve = [&](size_t bytes) { void* r = (void*)p; p += (bytes + 255) & ~(size_t)255; return r; };
  bf16_t* xb   = (bf16_t*)carve((size_t)M_ * D_  * 2);
  bf16_t* wqb  = (bf16_t*)carve((size_t)D_  * D_ * 2);
  bf16_t* wkb  = (bf16_t*)carve((size_t)KD_ * D_ * 2);
  bf16_t* wvb  = (bf16_t*)carve((size_t)KD_ * D_ * 2);
  bf16_t* wob  = (bf16_t*)carve((size_t)D_  * D_ * 2);
  float*  qf   = (float*)carve((size_t)M_ * D_  * 4);
  float*  kf   = (float*)carve((size_t)M_ * KD_ * 4);
  float*  vf   = (float*)carve((size_t)M_ * KD_ * 4);
  bf16_t* qbuf = (bf16_t*)carve((size_t)M_ * D_  * 2);
  bf16_t* kbuf = (bf16_t*)carve((size_t)M_ * KD_ * 2);
  bf16_t* vtb  = (bf16_t*)carve((size_t)M_ * KD_ * 2);
  bf16_t* obuf = (bf16_t*)carve((size_t)M_ * D_  * 2);

  auto cvt = [&](const float* src, bf16_t* dst, size_t n) {
    long n4 = (long)(n / 4);
    cvt_kernel<<<dim3((unsigned)((n4 + 255) / 256)), dim3(256), 0, stream>>>(src, dst, n4);
  };
  cvt(x,  xb,  (size_t)M_ * D_);
  cvt(wq, wqb, (size_t)D_  * D_);
  cvt(wk, wkb, (size_t)KD_ * D_);
  cvt(wv, wvb, (size_t)KD_ * D_);
  cvt(wo, wob, (size_t)D_  * D_);

  // projections
  gemm_bf16_kernel<<<dim3(D_ / BN,  M_ / BM), dim3(256), 0, stream>>>(xb, wqb, qf, M_, D_,  D_);
  gemm_bf16_kernel<<<dim3(KD_ / BN, M_ / BM), dim3(256), 0, stream>>>(xb, wkb, kf, M_, KD_, D_);
  gemm_bf16_kernel<<<dim3(KD_ / BN, M_ / BM), dim3(256), 0, stream>>>(xb, wvb, vf, M_, KD_, D_);

  // RoPE + repack
  rope_q_kernel<<<dim3((unsigned)((size_t)M_ * D_  / 256)), dim3(256), 0, stream>>>(qf, cosb, sinb, qbuf);
  rope_k_kernel<<<dim3((unsigned)((size_t)M_ * KD_ / 256)), dim3(256), 0, stream>>>(kf, cosb, sinb, kbuf);
  pack_v_kernel<<<dim3((unsigned)((size_t)M_ * KD_ / 256)), dim3(256), 0, stream>>>(vf, vtb);

  // flash attention: B_*H_*(S_/16) = 8192 waves, 8 waves/block
  attn_kernel<<<dim3(B_ * H_ * (S_ / 16) / 8), dim3(256), 0, stream>>>(qbuf, kbuf, vtb, obuf);

  // output projection -> f32 d_out
  gemm_bf16_kernel<<<dim3(D_ / BN, M_ / BM), dim3(256), 0, stream>>>(obuf, wob, out, M_, D_, D_);
}
